// ModalityAwareDualAttention_41300405518928
// MI455X (gfx1250) — matrix operand order repaired and checked by reading, not compiled, not run on hardware
//
#include <hip/hip_runtime.h>
#include <math.h>

typedef __attribute__((ext_vector_type(16))) __bf16 v16bf;
typedef __attribute__((ext_vector_type(8)))  float  v8f;

#define DEVINL __device__ __forceinline__

#ifndef __has_builtin
#define __has_builtin(x) 0
#endif

#if __has_builtin(__builtin_amdgcn_global_load_async_to_lds_b128)
#define HAVE_ASYNC_LDS 1
typedef int v4i_vs __attribute__((vector_size(16)));
typedef __attribute__((address_space(1))) v4i_vs as1_v4i;
typedef __attribute__((address_space(3))) v4i_vs as3_v4i;
#else
#define HAVE_ASYNC_LDS 0
#endif

template <int N>
DEVINL void waitAsync() {
#if __has_builtin(__builtin_amdgcn_s_wait_asynccnt)
  __builtin_amdgcn_s_wait_asynccnt((short)N);
#else
  if constexpr (N == 3) asm volatile("s_wait_asynccnt 0x3" ::: "memory");
  else                  asm volatile("s_wait_asynccnt 0x0" ::: "memory");
#endif
}

// Problem constants
constexpr int Bv = 32, Cv = 2048, Hv = 48, Wdim = 24, Pv = 3;
constexpr int ICv = 128, C4v = 512, PHv = 16, HDv = 8, WDv = 12, Nv = 96;
constexpr int BPv = Bv * Pv;            // 96
constexpr int KTC  = Cv / 32;           // 64 K-tiles over C
constexpr int NTN  = Nv / 16;           // 6  N-tiles over N=96
constexpr int MTIC = ICv / 16;          // 8  M-tiles for Q/K
constexpr int MTC  = Cv / 16;           // 128 M-tiles for V/AO
constexpr int KTJ  = Nv / 32;           // 3  K-tiles over j=96

// V-GEMM staging
constexpr int VCH    = 2;               // k-tiles per stage
constexpr int VNST   = KTC / VCH;       // 32 stages
constexpr int VSLICE = VCH * NTN * 512; // 6144 shorts = 12 KB per stage

DEVINL unsigned short f2bf(float f) {           // RNE float->bf16
  unsigned int u = __float_as_uint(f);
  u += 0x7FFFu + ((u >> 16) & 1u);
  return (unsigned short)(u >> 16);
}
// In-tile index for B fragment (32x16, K x col), 512 elems per tile
DEVINL int bswz(int K, int col) { int l = col + ((K & 16) ? 16 : 0); return l * 16 + (K & 15); }
// In-tile index for A fragment (16x32, m x K)
DEVINL int aswz(int m, int K) {
  int l = m + ((K & 16) ? 16 : 0);
  int h = K & 15;
  return l * 16 + h;
}
DEVINL float waveSum(float v) { for (int m = 16; m > 0; m >>= 1) v += __shfl_xor(v, m, 32); return v; }
DEVINL float waveMax(float v) { for (int m = 16; m > 0; m >>= 1) v = fmaxf(v, __shfl_xor(v, m, 32)); return v; }

DEVINL float bilin(const float* __restrict__ ao, int ph, int w) {
  float fy = ph * 0.5f - 0.25f, fx = w * 0.5f - 0.25f;
  int y0 = (int)floorf(fy), x0 = (int)floorf(fx);
  float wy = fy - (float)y0, wx = fx - (float)x0;
  int y1 = y0 + 1, x1 = x0 + 1;
  y0 = min(max(y0, 0), HDv - 1); y1 = min(max(y1, 0), HDv - 1);
  x0 = min(max(x0, 0), WDv - 1); x1 = min(max(x1, 0), WDv - 1);
  float v00 = ao[y0 * WDv + x0], v01 = ao[y0 * WDv + x1];
  float v10 = ao[y1 * WDv + x0], v11 = ao[y1 * WDv + x1];
  return (1.f - wy) * ((1.f - wx) * v00 + wx * v01) + wy * ((1.f - wx) * v10 + wx * v11);
}

// K1: downsample 2x2 -> xd, write bf16 B-fragment-swizzled activations
__global__ void k_prep_act(const float* __restrict__ x, unsigned short* __restrict__ act) {
  const long total = (long)BPv * Cv * Nv;
  for (long idx = (long)blockIdx.x * blockDim.x + threadIdx.x; idx < total;
       idx += (long)gridDim.x * blockDim.x) {
    int n = (int)(idx % Nv); long t = idx / Nv;
    int c = (int)(t % Cv); int bp = (int)(t / Cv);
    int b = bp / Pv, p = bp % Pv;
    int hd = n / WDv, wd = n % WDv;
    const float* s = x + (((long)(b * Cv + c) * Hv + (p * PHv + 2 * hd)) * Wdim + 2 * wd);
    float v = 0.25f * (s[0] + s[1] + s[Wdim] + s[Wdim + 1]);
    long off = (((long)bp * (KTC * NTN)) + (c >> 5) * NTN + (n >> 4)) * 512 + bswz(c & 31, n & 15);
    act[off] = f2bf(v);
  }
}

// K2: fold depthwise scale into Wq/Wk, bf16 A-fragment-swizzled
__global__ void k_fold_qk(const float* __restrict__ wq, const float* __restrict__ dwqw,
                          const float* __restrict__ wk, const float* __restrict__ dwkw,
                          unsigned short* __restrict__ wqs, unsigned short* __restrict__ wks) {
  const int total = 2 * Pv * ICv * Cv;
  for (int idx = blockIdx.x * blockDim.x + threadIdx.x; idx < total;
       idx += gridDim.x * blockDim.x) {
    int sel = idx / (Pv * ICv * Cv); int r = idx % (Pv * ICv * Cv);
    int p = r / (ICv * Cv); int r2 = r % (ICv * Cv);
    int i = r2 / Cv; int c = r2 % Cv;
    const float* ws = sel ? wk : wq; const float* ds = sel ? dwkw : dwqw;
    float v = ws[(p * ICv + i) * Cv + c] * ds[p * Cv + c];
    int off = ((p * MTIC + (i >> 4)) * KTC + (c >> 5)) * 512 + aswz(i & 15, c & 31);
    (sel ? wks : wqs)[off] = f2bf(v);
  }
}

// K2b: effective bias: pa_q_b + W @ dw_bias  (one wave per (sel,p,i))
__global__ void k_fold_bias(const float* __restrict__ wq, const float* __restrict__ qbd, const float* __restrict__ qb,
                            const float* __restrict__ wk, const float* __restrict__ kbd, const float* __restrict__ kb,
                            float* __restrict__ qbe, float* __restrict__ kbe) {
  int gid = blockIdx.x * (blockDim.x >> 5) + (threadIdx.x >> 5);
  int lane = threadIdx.x & 31;
  if (gid >= 2 * Pv * ICv) return;
  int sel = gid / (Pv * ICv); int r = gid % (Pv * ICv);
  int p = r / ICv; int i = r % ICv;
  const float* w = sel ? wk : wq; const float* bd = sel ? kbd : qbd;
  float s = 0.f;
  for (int c = lane; c < Cv; c += 32) s += w[(p * ICv + i) * Cv + c] * bd[p * Cv + c];
  s = waveSum(s);
  if (lane == 0) (sel ? kbe : qbe)[p * ICv + i] = s + (sel ? kb : qb)[p * ICv + i];
}

// K3: Wv -> bf16 A-fragment-swizzled
__global__ void k_swz_wv(const float* __restrict__ wv, unsigned short* __restrict__ wvs) {
  const int total = Pv * Cv * Cv;
  for (int idx = blockIdx.x * blockDim.x + threadIdx.x; idx < total;
       idx += gridDim.x * blockDim.x) {
    int p = idx / (Cv * Cv); int r = idx % (Cv * Cv);
    int d = r / Cv; int c = r % Cv;
    int off = ((p * MTC + (d >> 4)) * KTC + (c >> 5)) * 512 + aswz(d & 15, c & 31);
    wvs[off] = f2bf(wv[idx]);
  }
}

// K4: Q = Wq'*xd + bq'  and  K = Wk'*xd + bk'  (register double-buffered WMMA)
__global__ void __launch_bounds__(256)
k_gemm_qk(const unsigned short* __restrict__ wqs, const unsigned short* __restrict__ wks,
          const float* __restrict__ qbe, const float* __restrict__ kbe,
          const unsigned short* __restrict__ act,
          float* __restrict__ Qf, float* __restrict__ Kf) {
  int sel = blockIdx.x & 1; int bp = blockIdx.x >> 1;
  int mt = threadIdx.x >> 5, lane = threadIdx.x & 31;
  int p = bp % Pv;
  const unsigned short* W = sel ? wks : wqs;
  const float* be = sel ? kbe : qbe;
  float* Out = sel ? Kf : Qf;
  v8f acc[NTN];
#pragma unroll
  for (int i = 0; i < NTN; ++i) acc[i] = (v8f){0.f,0.f,0.f,0.f,0.f,0.f,0.f,0.f};
  const unsigned short* aB = W + ((p * MTIC + mt) * KTC) * 512 + lane * 16;
  const unsigned short* bB = act + ((long)bp * (KTC * NTN)) * 512 + lane * 16;

  v16bf aC = *(const v16bf*)(aB);
  v16bf bC[NTN];
#pragma unroll
  for (int nt = 0; nt < NTN; ++nt) bC[nt] = *(const v16bf*)(bB + nt * 512);

  for (int kt = 0; kt < KTC; ++kt) {
    int ktn = (kt + 1 < KTC) ? kt + 1 : kt;
    v16bf aN = *(const v16bf*)(aB + ktn * 512);
    v16bf bN[NTN];
#pragma unroll
    for (int nt = 0; nt < NTN; ++nt) bN[nt] = *(const v16bf*)(bB + (ktn * NTN + nt) * 512);
#pragma unroll
    for (int nt = 0; nt < NTN; ++nt)
      acc[nt] = __builtin_amdgcn_wmma_f32_16x16x32_bf16(false, aC, false, bC[nt], (short)0, acc[nt], false, false);
    aC = aN;
#pragma unroll
    for (int nt = 0; nt < NTN; ++nt) bC[nt] = bN[nt];
  }
  int mh = (lane >= 16) ? 8 : 0, col0 = lane & 15;
#pragma unroll
  for (int nt = 0; nt < NTN; ++nt) {
    int n = nt * 16 + col0;
#pragma unroll
    for (int r = 0; r < 8; ++r) {
      int i = mt * 16 + r + mh;
      Out[(long)(bp * ICv + i) * Nv + n] = acc[nt][r] + be[p * ICv + i];
    }
  }
}

// K5: V = Wv*xd + bv ; LDS-staged, double-buffered, async copy when available.
//     Output stored directly as bf16 A-fragments for the AO GEMM.
__global__ void __launch_bounds__(256)
k_gemm_v(const unsigned short* __restrict__ wvs, const float* __restrict__ vb,
         const unsigned short* __restrict__ act, unsigned short* __restrict__ vfrag) {
  __shared__ alignas(128) unsigned short ldsB[2][VSLICE];
  const int bp = blockIdx.x >> 4, mblk = blockIdx.x & 15;
  const int wave = threadIdx.x >> 5, lane = threadIdx.x & 31, tid = threadIdx.x;
  const int mt = mblk * 8 + wave, p = bp % Pv;

  v8f acc[NTN];
#pragma unroll
  for (int i = 0; i < NTN; ++i) acc[i] = (v8f){0.f,0.f,0.f,0.f,0.f,0.f,0.f,0.f};
  const unsigned short* aBase = wvs + ((p * MTC + mt) * KTC) * 512 + lane * 16;
  const unsigned short* gB = act + (size_t)bp * (KTC * NTN) * 512;

  v16bf aC = *(const v16bf*)(aBase);   // A fragment for kt = 0

#if HAVE_ASYNC_LDS
  {  // issue stage 0 copy (async global -> LDS)
    const unsigned short* s0 = gB + tid * 8;
    unsigned short* d0 = &ldsB[0][tid * 8];
#pragma unroll
    for (int j = 0; j < 3; ++j)
      __builtin_amdgcn_global_load_async_to_lds_b128(
          (as1_v4i*)(s0 + j * 2048), (as3_v4i*)(d0 + j * 2048), 0, 0);
  }
  for (int s = 0; s < VNST; ++s) {
    __syncthreads();                       // buf[(s+1)&1] free of readers
    if (s + 1 < VNST) {
      const unsigned short* sp = gB + (size_t)(s + 1) * VSLICE + tid * 8;
      unsigned short* dp = &ldsB[(s + 1) & 1][tid * 8];
#pragma unroll
      for (int j = 0; j < 3; ++j)
        __builtin_amdgcn_global_load_async_to_lds_b128(
            (as1_v4i*)(sp + j * 2048), (as3_v4i*)(dp + j * 2048), 0, 0);
      waitAsync<3>();                      // stage s copy complete (in-order)
    } else {
      waitAsync<0>();
    }
    __syncthreads();                       // all waves see stage s data
    const unsigned short* lb = &ldsB[s & 1][lane * 16];
#pragma unroll
    for (int kk = 0; kk < VCH; ++kk) {
      int kt = s * VCH + kk;
      int ktn = (kt + 1 < KTC) ? kt + 1 : kt;
      v16bf aN = *(const v16bf*)(aBase + ktn * 512);   // prefetch next A
#pragma unroll
      for (int nt = 0; nt < NTN; ++nt) {
        v16bf b = *(const v16bf*)(lb + (kk * NTN + nt) * 512);
        acc[nt] = __builtin_amdgcn_wmma_f32_16x16x32_bf16(false, aC, false, b, (short)0, acc[nt], false, false);
      }
      aC = aN;
    }
  }
#else
  {  // fallback: register-buffered copy + ds_store, same pipeline shape
    uint4 rg[3];
    const uint4* gB4 = (const uint4*)gB;
#pragma unroll
    for (int j = 0; j < 3; ++j) rg[j] = gB4[j * 256 + tid];   // stage 0
    for (int s = 0; s < VNST; ++s) {
      __syncthreads();
      uint4* lb4 = (uint4*)&ldsB[s & 1][0];
#pragma unroll
      for (int j = 0; j < 3; ++j) lb4[j * 256 + tid] = rg[j];
      if (s + 1 < VNST) {
#pragma unroll
        for (int j = 0; j < 3; ++j) rg[j] = gB4[(size_t)(s + 1) * (VSLICE / 8) + j * 256 + tid];
      }
      __syncthreads();
      const unsigned short* lb = &ldsB[s & 1][lane * 16];
#pragma unroll
      for (int kk = 0; kk < VCH; ++kk) {
        int kt = s * VCH + kk;
        int ktn = (kt + 1 < KTC) ? kt + 1 : kt;
        v16bf aN = *(const v16bf*)(aBase + ktn * 512);
#pragma unroll
        for (int nt = 0; nt < NTN; ++nt) {
          v16bf b = *(const v16bf*)(lb + (kk * NTN + nt) * 512);
          acc[nt] = __builtin_amdgcn_wmma_f32_16x16x32_bf16(false, aC, false, b, (short)0, acc[nt], false, false);
        }
        aC = aN;
      }
    }
  }
#endif

  int mh = (lane >= 16) ? 8 : 0, col0 = lane & 15;
#pragma unroll
  for (int nt = 0; nt < NTN; ++nt) {
#pragma unroll
    for (int r = 0; r < 8; ++r) {
      int m = r + mh;                       // row within 16-tile
      int d = mt * 16 + m;                  // output channel
      float val = acc[nt][r] + vb[p * Cv + d];
      int K = (nt & 1) * 16 + col0;         // j within 32-tile
      int off = (bp * (MTC * KTJ) + mt * KTJ + (nt >> 1)) * 512 + aswz(m, K);
      vfrag[off] = f2bf(val);
    }
  }
}

// K6: energy + softmax (one wave per (bp, row)); emit bf16 B-fragments of attn^T
__global__ void k_attn(const float* __restrict__ Qf, const float* __restrict__ Kf,
                       float* __restrict__ attn, unsigned short* __restrict__ attnf) {
  int gid = blockIdx.x * (blockDim.x >> 5) + (threadIdx.x >> 5);
  int lane = threadIdx.x & 31;
  if (gid >= BPv * Nv) return;
  int bp = gid / Nv, irow = gid % Nv;
  const float* qb = Qf + (long)bp * ICv * Nv;
  const float* kb = Kf + (long)bp * ICv * Nv;
  float a0 = 0.f, a1 = 0.f, a2 = 0.f;
  for (int i = 0; i < ICv; ++i) {
    float qv = qb[i * Nv + irow];
    const float* kr = kb + i * Nv;
    a0 += qv * kr[lane]; a1 += qv * kr[lane + 32]; a2 += qv * kr[lane + 64];
  }
  float mx = waveMax(fmaxf(a0, fmaxf(a1, a2)));
  float e0 = __expf(a0 - mx), e1 = __expf(a1 - mx), e2 = __expf(a2 - mx);
  float inv = 1.f / waveSum(e0 + e1 + e2);
  e0 *= inv; e1 *= inv; e2 *= inv;
  float* arow = attn + ((long)bp * Nv + irow) * Nv;
  arow[lane] = e0; arow[lane + 32] = e1; arow[lane + 64] = e2;
  unsigned short* afb = attnf + bp * (KTJ * NTN) * 512;
  int nt = irow >> 4, col = irow & 15;
  int m0 = lane, m1 = lane + 32, m2 = lane + 64;
  afb[((m0 >> 5) * NTN + nt) * 512 + bswz(m0 & 31, col)] = f2bf(e0);
  afb[((m1 >> 5) * NTN + nt) * 512 + bswz(m1 & 31, col)] = f2bf(e1);
  afb[((m2 >> 5) * NTN + nt) * 512 + bswz(m2 & 31, col)] = f2bf(e2);
}

// K7: AO = V * attn^T  (WMMA), f32 out
__global__ void __launch_bounds__(256)
k_gemm_ao(const unsigned short* __restrict__ vfrag, const unsigned short* __restrict__ attnf,
          float* __restrict__ aof) {
  int bp = blockIdx.x >> 4; int mblk = blockIdx.x & 15;
  int wave = threadIdx.x >> 5, lane = threadIdx.x & 31;
  int mt = mblk * 8 + wave;
  v8f acc[NTN];
#pragma unroll
  for (int i = 0; i < NTN; ++i) acc[i] = (v8f){0.f,0.f,0.f,0.f,0.f,0.f,0.f,0.f};
  const unsigned short* aB = vfrag + (bp * (MTC * KTJ) + mt * KTJ) * 512 + lane * 16;
  const unsigned short* bB = attnf + bp * (KTJ * NTN) * 512 + lane * 16;
#pragma unroll
  for (int kt = 0; kt < KTJ; ++kt) {
    v16bf a = *(const v16bf*)(aB + kt * 512);
#pragma unroll
    for (int nt = 0; nt < NTN; ++nt) {
      v16bf b = *(const v16bf*)(bB + (kt * NTN + nt) * 512);
      acc[nt] = __builtin_amdgcn_wmma_f32_16x16x32_bf16(false, a, false, b, (short)0, acc[nt], false, false);
    }
  }
  int mh = (lane >= 16) ? 8 : 0, col0 = lane & 15;
#pragma unroll
  for (int nt = 0; nt < NTN; ++nt)
#pragma unroll
    for (int r = 0; r < 8; ++r)
      aof[((long)bp * Cv + mt * 16 + r + mh) * Nv + nt * 16 + col0] = acc[nt][r];
}

// K8: gap[bp,c] = mean over (ph,w) of (pa_gamma*up + xp)
__global__ void k_gap(const float* __restrict__ aof, const float* __restrict__ x,
                      const float* __restrict__ pa_gamma, float* __restrict__ gap) {
  int bp = blockIdx.x / Cv, c = blockIdx.x % Cv;
  int b = bp / Pv, p = bp % Pv;
  const float* ao = aof + ((long)bp * Cv + c) * Nv;
  float g = pa_gamma[p];
  float s = 0.f;
  for (int sp = threadIdx.x; sp < PHv * Wdim; sp += blockDim.x) {
    int ph = sp / Wdim, w = sp % Wdim;
    float xpv = x[((long)(b * Cv + c) * Hv + p * PHv + ph) * Wdim + w];
    s += g * bilin(ao, ph, w) + xpv;
  }
  __shared__ float red[128];
  red[threadIdx.x] = s; __syncthreads();
  for (int st = 64; st > 0; st >>= 1) {
    if ((int)threadIdx.x < st) red[threadIdx.x] += red[threadIdx.x + st];
    __syncthreads();
  }
  if (threadIdx.x == 0) gap[bp * Cv + c] = red[0] * (1.f / (PHv * Wdim));
}

// K9: h1 = relu(fc1_w @ gap + b)   (wave per (bp,d))
__global__ void k_fc1(const float* __restrict__ gap, const float* __restrict__ fw,
                      const float* __restrict__ fb, float* __restrict__ h1) {
  int gid = blockIdx.x * (blockDim.x >> 5) + (threadIdx.x >> 5);
  int lane = threadIdx.x & 31;
  if (gid >= BPv * C4v) return;
  int bp = gid / C4v, d = gid % C4v; int p = bp % Pv;
  float s = 0.f;
  for (int c = lane; c < Cv; c += 32) s += gap[bp * Cv + c] * fw[(long)(p * C4v + d) * Cv + c];
  s = waveSum(s);
  if (lane == 0) h1[bp * C4v + d] = fmaxf(s + fb[p * C4v + d], 0.f);
}

// K10: cw = sigmoid(fc2_w @ h1 + b)   (wave per (bp,c))
__global__ void k_fc2(const float* __restrict__ h1, const float* __restrict__ fw,
                      const float* __restrict__ fb, float* __restrict__ cw) {
  int gid = blockIdx.x * (blockDim.x >> 5) + (threadIdx.x >> 5);
  int lane = threadIdx.x & 31;
  if (gid >= BPv * Cv) return;
  int bp = gid / Cv, c = gid % Cv; int p = bp % Pv;
  float s = 0.f;
  for (int d = lane; d < C4v; d += 32) s += h1[bp * C4v + d] * fw[(long)(p * Cv + c) * C4v + d];
  s = waveSum(s);
  if (lane == 0) cw[bp * Cv + c] = 1.f / (1.f + __expf(-(s + fb[p * Cv + c])));
}

// K11: modality gate MLP
__global__ void k_gate(const int* __restrict__ modality, const float* __restrict__ w1,
                       const float* __restrict__ b1, const float* __restrict__ w2,
                       const float* __restrict__ b2, float* __restrict__ mw) {
  int t = threadIdx.x;
  if (t >= Bv * Pv) return;
  int b = t / Pv, pp = t % Pv;
  float mf = (float)modality[b];
  float s = 0.f;
  for (int j = 0; j < 12; ++j) {
    float g = fmaxf(mf * w1[j] + b1[j], 0.f);
    s += g * w2[pp * 12 + j];
  }
  mw[t] = 1.f / (1.f + __expf(-(s + b2[pp])));
}

// K12: upsample + residual + channel attn + modality blend -> output
__global__ void k_final(const float* __restrict__ x, const float* __restrict__ aof,
                        const float* __restrict__ cw, const float* __restrict__ mw,
                        const float* __restrict__ pa_gamma, const float* __restrict__ ca_gamma,
                        float* __restrict__ out) {
  const long total = (long)Bv * Cv * Hv * Wdim;
  for (long idx = (long)blockIdx.x * blockDim.x + threadIdx.x; idx < total;
       idx += (long)gridDim.x * blockDim.x) {
    int w = (int)(idx % Wdim); long t = idx / Wdim;
    int h = (int)(t % Hv); t /= Hv;
    int c = (int)(t % Cv); int b = (int)(t / Cv);
    int p = h / PHv, ph = h % PHv; int bp = b * Pv + p;
    const float* ao = aof + ((long)bp * Cv + c) * Nv;
    float xpv = x[idx];
    float pa  = pa_gamma[p] * bilin(ao, ph, w) + xpv;
    float caf = pa * (1.f + ca_gamma[p] * cw[bp * Cv + c]);
    float wg  = mw[bp];
    out[idx] = xpv * (1.f - wg) + caf * wg;
  }
}

extern "C" void kernel_launch(void* const* d_in, const int* in_sizes, int n_in,
                              void* d_out, int out_size, void* d_ws, size_t ws_size,
                              hipStream_t stream) {
  const float* x      = (const float*)d_in[0];
  const float* dwq_w  = (const float*)d_in[1];
  const float* dwq_b  = (const float*)d_in[2];
  const float* q_w    = (const float*)d_in[3];
  const float* q_b    = (const float*)d_in[4];
  const float* dwk_w  = (const float*)d_in[5];
  const float* dwk_b  = (const float*)d_in[6];
  const float* k_w    = (const float*)d_in[7];
  const float* k_b    = (const float*)d_in[8];
  const float* v_w    = (const float*)d_in[9];
  const float* v_b    = (const float*)d_in[10];
  const float* pa_g   = (const float*)d_in[11];
  const float* fc1_w  = (const float*)d_in[12];
  const float* fc1_b  = (const float*)d_in[13];
  const float* fc2_w  = (const float*)d_in[14];
  const float* fc2_b  = (const float*)d_in[15];
  const float* ca_g   = (const float*)d_in[16];
  const float* gw1    = (const float*)d_in[17];
  const float* gb1    = (const float*)d_in[18];
  const float* gw2    = (const float*)d_in[19];
  const float* gb2    = (const float*)d_in[20];
  const int*   moda   = (const int*)d_in[21];
  float* out = (float*)d_out;

  auto align256 = [](size_t v) { return (v + 255) & ~(size_t)255; };
  char* ws = (char*)d_ws;
  size_t off = 0;
  auto take = [&](size_t bytes) { char* p = ws + off; off = align256(off + bytes); return p; };

  unsigned short* ACT   = (unsigned short*)take((size_t)BPv * Cv * Nv * 2);   // 37.7 MB
  unsigned short* WQS   = (unsigned short*)take((size_t)Pv * ICv * Cv * 2);   // 1.57 MB
  unsigned short* WKS   = (unsigned short*)take((size_t)Pv * ICv * Cv * 2);
  float*          QBE   = (float*)take((size_t)Pv * ICv * 4);
  float*          KBE   = (float*)take((size_t)Pv * ICv * 4);
  unsigned short* WVS   = (unsigned short*)take((size_t)Pv * Cv * Cv * 2);    // 25.2 MB
  float*          QF    = (float*)take((size_t)BPv * ICv * Nv * 4);           // 4.7 MB
  float*          KF    = (float*)take((size_t)BPv * ICv * Nv * 4);
  unsigned short* VFRAG = (unsigned short*)take((size_t)BPv * Cv * Nv * 2);   // 37.7 MB
  float*          ATTN  = (float*)take((size_t)BPv * Nv * Nv * 4);            // 3.5 MB
  unsigned short* ATTNF = (unsigned short*)take((size_t)BPv * Nv * Nv * 2);
  float*          AOF   = (float*)take((size_t)BPv * Cv * Nv * 4);            // 75.5 MB
  float*          GAP   = (float*)take((size_t)BPv * Cv * 4);
  float*          H1    = (float*)take((size_t)BPv * C4v * 4);
  float*          CW    = (float*)take((size_t)BPv * Cv * 4);
  float*          MW    = (float*)take((size_t)BPv * 4);
  (void)ws_size; (void)in_sizes; (void)n_in; (void)out_size;

  // 1. activations (downsample + swizzle)
  k_prep_act<<<73728, 256, 0, stream>>>(x, ACT);
  // 2. weight folding / swizzling
  k_fold_qk<<<6144, 256, 0, stream>>>(q_w, dwq_w, k_w, dwk_w, WQS, WKS);
  k_fold_bias<<<96, 256, 0, stream>>>(q_w, dwq_b, q_b, k_w, dwk_b, k_b, QBE, KBE);
  k_swz_wv<<<49152, 256, 0, stream>>>(v_w, WVS);
  // 3. projections (WMMA)
  k_gemm_qk<<<BPv * 2, 256, 0, stream>>>(WQS, WKS, QBE, KBE, ACT, QF, KF);
  k_gemm_v<<<BPv * 16, 256, 0, stream>>>(WVS, v_b, ACT, VFRAG);
  // 4. attention
  k_attn<<<(BPv * Nv + 7) / 8, 256, 0, stream>>>(QF, KF, ATTN, ATTNF);
  k_gemm_ao<<<BPv * 16, 256, 0, stream>>>(VFRAG, ATTNF, AOF);
  // 5. channel attention path
  k_gap<<<BPv * Cv, 128, 0, stream>>>(AOF, x, pa_g, GAP);
  k_fc1<<<(BPv * C4v + 7) / 8, 256, 0, stream>>>(GAP, fc1_w, fc1_b, H1);
  k_fc2<<<(BPv * Cv + 7) / 8, 256, 0, stream>>>(H1, fc2_w, fc2_b, CW);
  // 6. modality gate + final blend
  k_gate<<<1, 128, 0, stream>>>(moda, gw1, gb1, gw2, gb2, MW);
  k_final<<<294912, 256, 0, stream>>>(x, AOF, CW, MW, pa_g, ca_g, out);
}
